// TGAP_16458314678747
// MI455X (gfx1250) — matrix operand order repaired and checked by reading, not compiled, not run on hardware
//
#include <hip/hip_runtime.h>
#include <math.h>

// Problem constants (match reference)
#define N_NODES 5000
#define E_EDGES 16000
#define B_Q     8
#define D_DIM   128
#define DD_DIM  64
#define H4D     512   // 4*D
#define H2D     256   // 2*D

typedef __bf16 bf16_t;
typedef __attribute__((ext_vector_type(16))) __bf16 v16bf;
typedef __attribute__((ext_vector_type(8)))  float  v8f;
typedef __attribute__((ext_vector_type(4)))  unsigned int v4u;
typedef __attribute__((ext_vector_type(8)))  int v8i;
typedef __attribute__((ext_vector_type(4)))  int v4i;

union FragU { uint4 q[2]; v16bf v; };

// ---- WMMA fragment loaders (CDNA5 bf16 16x16x32 layouts, ISA 7.12.2) ----
// A matrix 16x32 (MxK), row-major source with leading dim ld (elements).
// lane L: M = L&15 ; K-chunks at 8*(L>>4) and 16+8*(L>>4), 8 elems each.
__device__ __forceinline__ v16bf frag_a_load(const bf16_t* base, int ld) {
  const int lane = threadIdx.x & 31;
  const int m    = lane & 15;
  const int kb   = (lane >> 4) << 3;           // 0 or 8
  const bf16_t* p = base + m * ld + kb;
  FragU u;
  u.q[0] = *(const uint4*)(p);                 // K = kb .. kb+7
  u.q[1] = *(const uint4*)(p + 16);            // K = kb+16 .. kb+23
  return u.v;
}

// B matrix 32x16 (KxN), loaded from PRE-TRANSPOSED weights Wt (NxK row-major).
// lane L: N = L&15 ; K-chunks at 8*(L>>4) and 16+8*(L>>4), 8 elems each.
__device__ __forceinline__ v16bf frag_b_load(const bf16_t* baseT, int ld) {
  const int lane = threadIdx.x & 31;
  const int n    = lane & 15;
  const int kb   = (lane >> 4) << 3;           // 0 or 8
  const bf16_t* p = baseT + n * ld + kb;
  FragU u;
  u.q[0] = *(const uint4*)(p);
  u.q[1] = *(const uint4*)(p + 16);
  return u.v;
}

__device__ __forceinline__ v8f wmma_bf16(v16bf a, v16bf b, v8f c) {
  return __builtin_amdgcn_wmma_f32_16x16x32_bf16(
      /*neg_a=*/false, a, /*neg_b=*/false, b,
      /*c_mod=*/(short)0, c, /*reuse_a=*/false, /*reuse_b=*/false);
}

__device__ __forceinline__ void atomicMaxFloat(float* addr, float val) {
  if (val >= 0.f) atomicMax((int*)addr, __float_as_int(val));
  else            atomicMin((unsigned int*)addr, __float_as_uint(val));
}

__device__ __forceinline__ float leaky(float x) { return x > 0.f ? x : 0.01f * x; }

// ---------------------------------------------------------------------------
// 0) init softmax stats: m = -inf, z = 0, num = 0
__global__ void init_stats_kernel(float* m, float* z, float* num, int n) {
  int i = blockIdx.x * blockDim.x + threadIdx.x;
  if (i < n) { m[i] = -INFINITY; z[i] = 0.f; num[i] = 0.f; }
}

// 1) diachronic node embedding h_n (N x D, f32)
__global__ void hn_kernel(const int* node_idx, const float* node_time,
                          const float* syn, const float* dia,
                          const float* dw, const float* db, float* hn) {
  int i = blockIdx.x * blockDim.x + threadIdx.x;
  if (i >= N_NODES * D_DIM) return;
  int nn = i >> 7, d = i & 127;
  int nid = node_idx[nn];
  float v = syn[nid * D_DIM + d];
  if (d >= DD_DIM) {
    int dd = d - DD_DIM;
    v += dia[nid * DD_DIM + dd] *
         sinf(dw[nid * DD_DIM + dd] * node_time[nn] + db[nid * DD_DIM + dd]);
  }
  hn[i] = v;
}

// 2) convert + transpose weights to bf16 (out-major so B-frags are contiguous)
__global__ void conv_w_kernel(const float* W1, const float* W2, const float* W3,
                              const float* Wss, const float* Wee,
                              bf16_t* W1t, bf16_t* W2t, bf16_t* W3t,
                              bf16_t* Wsst, bf16_t* Weet) {
  int i = blockIdx.x * blockDim.x + threadIdx.x;
  if (i < 512 * 128) {            // W1 top half: (128,512) -> (512,128)
    int o = i >> 7, ii = i & 127;
    W1t[i] = (bf16_t)W1[ii * H4D + o];  return;
  }
  i -= 512 * 128;
  if (i < 256 * 512) {            // W2: (512,256) -> (256,512)
    int o = i >> 9, ii = i & 511;
    W2t[i] = (bf16_t)W2[ii * H2D + o];  return;
  }
  i -= 256 * 512;
  if (i < 128 * 256) {            // W3: (256,128) -> (128,256)
    int o = i >> 8, ii = i & 255;
    W3t[i] = (bf16_t)W3[ii * D_DIM + o];  return;
  }
  i -= 128 * 256;
  if (i < 3 * 128 * 128) {        // W_ss[k]: (128,128) -> transposed
    int k = i >> 14, r = i & 16383, o = r >> 7, ii = r & 127;
    Wsst[i] = (bf16_t)Wss[(k * D_DIM + ii) * D_DIM + o];  return;
  }
  i -= 3 * 128 * 128;
  if (i < 3 * 128 * 128) {
    int k = i >> 14, r = i & 16383, o = r >> 7, ii = r & 127;
    Weet[i] = (bf16_t)Wee[(k * D_DIM + ii) * D_DIM + o];  return;
  }
}

// 3) batch half of layer1 (+b1 folded):  bp[b][o] = b1[o] + batch_t2v[b] . W1_bot[:,o]
__global__ void batch_part_kernel(const float* bt2v, const float* W1,
                                  const float* b1, float* bp) {
  int i = blockIdx.x * blockDim.x + threadIdx.x;
  if (i >= B_Q * H4D) return;
  int b = i >> 9, o = i & 511;
  float acc = b1[o];
  for (int ii = 0; ii < D_DIM; ++ii)
    acc += bt2v[b * D_DIM + ii] * W1[(D_DIM + ii) * H4D + o];
  bp[i] = acc;
}

// 4) edge half of layer1: edge_part (E x 512) = edge_t2v (E x 128) @ W1_top
__global__ __launch_bounds__(256) void edge_l1_kernel(const float* et2v,
                                                      const bf16_t* W1t,
                                                      float* epart) {
  __shared__ alignas(32) bf16_t sA[32][128];
  const int e0  = blockIdx.x * 32;
  const int tid = threadIdx.x;
  for (int i = tid; i < 32 * 128; i += 256)
    sA[i >> 7][i & 127] = (bf16_t)et2v[(e0 + (i >> 7)) * D_DIM + (i & 127)];
  __syncthreads();

  const int wave = tid >> 5, lane = tid & 31;
  const int n = lane & 15, hi = lane >> 4;
  for (int t = wave; t < 64; t += 8) {      // 2 M-tiles x 32 N-tiles
    int mt = t >> 5, nt = t & 31;
    v8f c = {};
    for (int kk = 0; kk < 4; ++kk) {
      v16bf a = frag_a_load(&sA[mt * 16][kk * 32], 128);
      v16bf b = frag_b_load(W1t + (nt * 16) * 128 + kk * 32, 128);
      c = wmma_bf16(a, b, c);
    }
    for (int r = 0; r < 8; ++r) {
      int m = mt * 16 + r + 8 * hi;
      epart[(size_t)(e0 + m) * H4D + nt * 16 + n] = c[r];
    }
  }
}

// 5) fused: relu(L1) -> L2 -> L3 -> tau -> masked lin3 GEMMs -> g, score-max
//    Dynamic LDS layout (bytes):
//      [0      , 32768)  sH1 (32x512 bf16)   -- later reused as sXA/sXB
//      [32768  , 49152)  sH2 (32x256 bf16)
//      [49152  , 114688) sW3 (128x256 bf16)  -- staged by TDM tensor_load_to_lds
#define SM_H2 (32 * H4D * 2)
#define SM_W3 (SM_H2 + 32 * H2D * 2)
#define SM_TOTAL (SM_W3 + D_DIM * H2D * 2)

__global__ __launch_bounds__(256) void fused_kernel(
    const int* src, const int* dst, const int* rel,
    const int* e_start, const int* e_end, const int* b_start, const int* b_end,
    const float* hn, const float* rel_embed,
    const float* epart, const float* bp,
    const bf16_t* W2t, const float* b2,
    const bf16_t* W3t, const float* b3,
    const bf16_t* Wsst, const float* bss,
    const bf16_t* Weet, const float* bee,
    const float* attn_i, const float* attn_j,
    int nblocks, bf16_t* gbuf, float* mbuf) {
  extern __shared__ __align__(32) char smem[];
  bf16_t (*sH1)[H4D]   = (bf16_t(*)[H4D])smem;
  bf16_t (*sH2)[H2D]   = (bf16_t(*)[H2D])(smem + SM_H2);
  bf16_t*  sW3         = (bf16_t*)(smem + SM_W3);
  bf16_t (*sXA)[D_DIM] = (bf16_t(*)[D_DIM])smem;               // aliases sH1
  bf16_t (*sXB)[D_DIM] = (bf16_t(*)[D_DIM])(smem + 32 * D_DIM * 2);
  __shared__ int sIdxSS[32], sIdxEE[32], sSrc[4], sDst[4], sRel[4];

  const int e0  = blockIdx.x * 4;       // 4 edges x 8 queries = 32 rows
  const int tid = threadIdx.x;

  // ---- TDM: DMA W3t (128 rows x 256 cols, bf16) into LDS (ISA ch.8 D#) ----
  if (tid == 0) {
    unsigned lds_off = (unsigned)(uintptr_t)(void*)sW3;  // low 32b of generic = LDS offset
    unsigned long long ga = (unsigned long long)(uintptr_t)W3t;
    v4u g0 = { 1u,                                   // count=1, user descriptor
               lds_off,                              // lds_addr [63:32]
               (unsigned)(ga & 0xffffffffu),         // global_addr low
               (unsigned)((ga >> 32) & 0x1ffffffu) | (2u << 30) };  // addr hi | type=2
    v8i g1 = { 0x00010000,                 // workgroup_mask=0, data_size=1 (2 bytes)
               (int)(256u << 16),          // tensor_dim0 = 256  (bits 79:48)
               (int)(128u << 16),          // tensor_dim1 = 128  (bits 111:80)
               (int)(256u << 16),          // tile_dim0   = 256  (bits 127:112)
               128,                        // tile_dim1   = 128  (bits 143:128)
               256,                        // tensor_dim0_stride = 256 (bits 207:160)
               0, 0 };
    v4i gz4 = { 0, 0, 0, 0 };              // 2-D tensor: groups 2/3 unused
    v8i gz8 = { 0, 0, 0, 0, 0, 0, 0, 0 };
    __builtin_amdgcn_tensor_load_to_lds(g0, g1, gz4, gz4, gz8, 0);
  }

  if (tid < 32) {
    int el = tid >> 3, b = tid & 7, eg = e0 + el;
    int ds = e_start[eg] - b_start[b];
    int de = e_end[eg]   - b_end[b];
    sIdxSS[tid] = (ds > 0) - (ds < 0) + 1;
    sIdxEE[tid] = (de > 0) - (de < 0) + 1;
    if (b == 0) { sSrc[el] = src[eg]; sDst[el] = dst[eg]; sRel[el] = rel[eg]; }
  }
  // stage H1 = relu(edge_part[e] + batch_part[b])   (b1 folded into bp)
  for (int i = tid; i < 32 * H4D; i += 256) {
    int r = i >> 9, c = i & 511, el = r >> 3, b = r & 7;
    float v = epart[(size_t)(e0 + el) * H4D + c] + bp[b * H4D + c];
    sH1[r][c] = (bf16_t)fmaxf(v, 0.f);
  }
  if (blockIdx.x + 1 < (unsigned)nblocks && tid < 64)  // gfx1250 global_prefetch
    __builtin_prefetch(&epart[(size_t)(e0 + 4) * H4D + tid * 32], 0, 1);
  __builtin_amdgcn_s_wait_tensorcnt(0);   // no-op in waves that issued nothing
  __syncthreads();

  const int wave = tid >> 5, lane = tid & 31;
  const int n = lane & 15, hi = lane >> 4;

  // ---- layer 2: (32x512)@(512x256), relu -> H2 bf16 ----
  for (int t = wave; t < 32; t += 8) {
    int mt = t >> 4, nt = t & 15;
    v8f c = {};
    for (int kk = 0; kk < 16; ++kk) {
      v16bf a = frag_a_load(&sH1[mt * 16][kk * 32], H4D);
      v16bf b = frag_b_load(W2t + (nt * 16) * H4D + kk * 32, H4D);
      c = wmma_bf16(a, b, c);
    }
    float bias = b2[nt * 16 + n];
    for (int r = 0; r < 8; ++r)
      sH2[mt * 16 + r + 8 * hi][nt * 16 + n] = (bf16_t)fmaxf(c[r] + bias, 0.f);
  }
  __syncthreads();

  // ---- layer 3 (B-frags from TDM-staged LDS) -> tau ; XA=hn[src]+tau, XB=rel+tau ----
  for (int t = wave; t < 16; t += 8) {
    int mt = t >> 3, nt = t & 7;
    v8f c = {};
    for (int kk = 0; kk < 8; ++kk) {
      v16bf a = frag_a_load(&sH2[mt * 16][kk * 32], H2D);
      v16bf b = frag_b_load(sW3 + (nt * 16) * H2D + kk * 32, H2D);
      c = wmma_bf16(a, b, c);
    }
    float bias = b3[nt * 16 + n];
    for (int r = 0; r < 8; ++r) {
      int m = mt * 16 + r + 8 * hi, el = m >> 3, d = nt * 16 + n;
      float tau = c[r] + bias;
      sXA[m][d] = (bf16_t)(hn[sSrc[el] * D_DIM + d] + tau);
      sXB[m][d] = (bf16_t)(rel_embed[sRel[el] * D_DIM + d] + tau);
    }
  }
  __syncthreads();

  // ---- lin3 as 3 row-masked GEMMs per branch; epilogue: g, score, atomic max ----
  for (int t = wave; t < 16; t += 8) {
    int mt = t >> 3, nt = t & 7;
    int rowA = mt * 16 + (lane & 15);       // A-frag row owned by this lane
    int iss = sIdxSS[rowA], iee = sIdxEE[rowA];
    v16bf aA[4], aB[4];
    for (int kk = 0; kk < 4; ++kk) {
      aA[kk] = frag_a_load(&sXA[mt * 16][kk * 32], D_DIM);
      aB[kk] = frag_a_load(&sXB[mt * 16][kk * 32], D_DIM);
    }
    const int d = nt * 16 + n;
    v8f c = {};
    for (int r = 0; r < 8; ++r) {           // per-row selected biases
      int m = mt * 16 + r + 8 * hi;
      c[r] = bss[sIdxSS[m] * D_DIM + d] + bee[sIdxEE[m] * D_DIM + d];
    }
    v16bf vz = {};
    for (int k = 0; k < 3; ++k) {
      for (int kk = 0; kk < 4; ++kk) {
        v16bf am = (iss == k) ? aA[kk] : vz;
        v16bf bm = frag_b_load(Wsst + k * D_DIM * D_DIM + (nt * 16) * D_DIM + kk * 32, D_DIM);
        c = wmma_bf16(am, bm, c);
      }
      for (int kk = 0; kk < 4; ++kk) {
        v16bf am = (iee == k) ? aB[kk] : vz;
        v16bf bm = frag_b_load(Weet + k * D_DIM * D_DIM + (nt * 16) * D_DIM + kk * 32, D_DIM);
        c = wmma_bf16(am, bm, c);
      }
    }
    float aj = attn_j[d], ain = attn_i[d];
    for (int r = 0; r < 8; ++r) {
      int m = mt * 16 + r + 8 * hi, el = m >> 3, b = m & 7, eg = e0 + el;
      float g = c[r];
      gbuf[((size_t)eg * B_Q + b) * D_DIM + d] = (bf16_t)g;
      float sc = leaky(hn[sDst[el] * D_DIM + d] * ain * (g * aj));
      atomicMaxFloat(&mbuf[((size_t)sDst[el] * B_Q + b) * D_DIM + d], sc);
    }
  }
}

// 6) pass 2: s = exp(score - m[dst]); accumulate z and numerator
__global__ void pass2_kernel(const int* dst, const bf16_t* gbuf, const float* hn,
                             const float* attn_i, const float* attn_j,
                             const float* mbuf, float* zbuf, float* nbuf) {
  int i = blockIdx.x * blockDim.x + threadIdx.x;
  if (i >= E_EDGES * B_Q * D_DIM) return;
  int d = i & 127, b = (i >> 7) & 7, e = i >> 10;
  int dn = dst[e];
  float g  = (float)gbuf[i];
  float sc = leaky(hn[dn * D_DIM + d] * attn_i[d] * (g * attn_j[d]));
  int oi = (dn * B_Q + b) * D_DIM + d;
  float s = __expf(sc - mbuf[oi]);
  atomicAdd(&zbuf[oi], s);
  atomicAdd(&nbuf[oi], s * g);
}

// 7) finalize: out = leaky_relu(num / (z + 1e-16))
__global__ void finalize_kernel(const float* zbuf, const float* nbuf,
                                float* out, int n) {
  int i = blockIdx.x * blockDim.x + threadIdx.x;
  if (i < n) out[i] = leaky(nbuf[i] / (zbuf[i] + 1e-16f));
}

// ---------------------------------------------------------------------------
extern "C" void kernel_launch(void* const* d_in, const int* in_sizes, int n_in,
                              void* d_out, int out_size, void* d_ws, size_t ws_size,
                              hipStream_t stream) {
  const int*   src       = (const int*)d_in[0];
  const int*   dst       = (const int*)d_in[1];
  const int*   node_idx  = (const int*)d_in[2];
  const int*   rel_type  = (const int*)d_in[3];
  const int*   e_start   = (const int*)d_in[4];
  const int*   e_end     = (const int*)d_in[5];
  const int*   b_start   = (const int*)d_in[6];
  const int*   b_end     = (const int*)d_in[7];
  const float* node_time = (const float*)d_in[8];
  const float* edge_t2v  = (const float*)d_in[9];
  const float* batch_t2v = (const float*)d_in[10];
  const float* syn_embed = (const float*)d_in[11];
  const float* dia_embed = (const float*)d_in[12];
  const float* dia_w     = (const float*)d_in[13];
  const float* dia_b     = (const float*)d_in[14];
  const float* rel_embed = (const float*)d_in[15];
  const float* ts_W1     = (const float*)d_in[16];
  const float* ts_b1     = (const float*)d_in[17];
  const float* ts_W2     = (const float*)d_in[18];
  const float* ts_b2     = (const float*)d_in[19];
  const float* ts_W3     = (const float*)d_in[20];
  const float* ts_b3     = (const float*)d_in[21];
  const float* W_ss      = (const float*)d_in[22];
  const float* b_ss      = (const float*)d_in[23];
  const float* W_ee      = (const float*)d_in[24];
  const float* b_ee      = (const float*)d_in[25];
  const float* attn_i    = (const float*)d_in[26];
  const float* attn_j    = (const float*)d_in[27];
  float* out = (float*)d_out;

  char* ws = (char*)d_ws;
  size_t off = 0;
  auto alloc = [&](size_t bytes) -> void* {
    void* p = ws + off;
    off += (bytes + 255) & ~(size_t)255;
    return p;
  };
  float*  hn    = (float*) alloc((size_t)N_NODES * D_DIM * 4);
  bf16_t* W1t   = (bf16_t*)alloc((size_t)H4D * D_DIM * 2);
  bf16_t* W2t   = (bf16_t*)alloc((size_t)H2D * H4D * 2);
  bf16_t* W3t   = (bf16_t*)alloc((size_t)D_DIM * H2D * 2);
  bf16_t* Wsst  = (bf16_t*)alloc((size_t)3 * D_DIM * D_DIM * 2);
  bf16_t* Weet  = (bf16_t*)alloc((size_t)3 * D_DIM * D_DIM * 2);
  float*  bp    = (float*) alloc((size_t)B_Q * H4D * 4);
  float*  epart = (float*) alloc((size_t)E_EDGES * H4D * 4);
  bf16_t* gbuf  = (bf16_t*)alloc((size_t)E_EDGES * B_Q * D_DIM * 2);
  float*  mbuf  = (float*) alloc((size_t)N_NODES * B_Q * D_DIM * 4);
  float*  zbuf  = (float*) alloc((size_t)N_NODES * B_Q * D_DIM * 4);
  float*  nbuf  = (float*) alloc((size_t)N_NODES * B_Q * D_DIM * 4);

  const int NBD = N_NODES * B_Q * D_DIM;            // 5,120,000
  init_stats_kernel<<<(NBD + 255) / 256, 256, 0, stream>>>(mbuf, zbuf, nbuf, NBD);
  hn_kernel<<<(N_NODES * D_DIM + 255) / 256, 256, 0, stream>>>(
      node_idx, node_time, syn_embed, dia_embed, dia_w, dia_b, hn);
  const int CONV = 512 * 128 + 256 * 512 + 128 * 256 + 2 * 3 * 128 * 128;
  conv_w_kernel<<<(CONV + 255) / 256, 256, 0, stream>>>(
      ts_W1, ts_W2, ts_W3, W_ss, W_ee, W1t, W2t, W3t, Wsst, Weet);
  batch_part_kernel<<<(B_Q * H4D + 255) / 256, 256, 0, stream>>>(
      batch_t2v, ts_W1, ts_b1, bp);
  edge_l1_kernel<<<E_EDGES / 32, 256, 0, stream>>>(edge_t2v, W1t, epart);
  const int FB = E_EDGES / 4;
  fused_kernel<<<FB, 256, SM_TOTAL, stream>>>(
      src, dst, rel_type, e_start, e_end, b_start, b_end,
      hn, rel_embed, epart, bp, W2t, ts_b2, W3t, ts_b3,
      Wsst, b_ss, Weet, b_ee, attn_i, attn_j, FB, gbuf, mbuf);
  pass2_kernel<<<(E_EDGES * B_Q * D_DIM + 255) / 256, 256, 0, stream>>>(
      dst, gbuf, hn, attn_i, attn_j, mbuf, zbuf, nbuf);
  finalize_kernel<<<(NBD + 255) / 256, 256, 0, stream>>>(zbuf, nbuf, out, NBD);
}
